// MEGNet_44641890075079
// MI455X (gfx1250) — matrix operand order repaired and checked by reading, not compiled, not run on hardware
//
// MEGNet forward for MI455X (gfx1250). Compile-only iteration.
// Strategy: f32 activations in HBM (traffic-bound), bf16 only at the WMMA
// boundary. All GEMM layers fused per 16-row tile through LDS using
// v_wmma_f32_16x16x32_bf16 (one wave32 per tile). Weights pre-cast once to a
// bf16 arena (<1MB, L2-resident). Segment reductions via f32 global atomics.
#include <hip/hip_runtime.h>
#include <stdint.h>
#include <stddef.h>

namespace {

constexpr int kNodes  = 50000;
constexpr int kEdges  = 800000;
constexpr int kGraphs = 512;
constexpr int kNH = 64;
constexpr int kEH = 64;
constexpr int kGH = 32;

typedef unsigned short bfr;
typedef __attribute__((ext_vector_type(16))) __bf16 v16bf;
typedef __attribute__((ext_vector_type(8)))  __bf16 v8bf;
typedef __attribute__((ext_vector_type(8)))  float  v8f;

struct Seg   { const float* p; const int* idx; int w; };
struct Layer { const bfr* W; const float* b; int N; int act; };
struct MlpArgs {
  Seg   s[4];
  int   nseg, K0, K0pad;
  Layer L[3];
  int   nl;
  long long M;
  float* out;
  int ld_out;
};

__device__ __forceinline__ bfr f2bf(float f) {
  unsigned int u = __float_as_uint(f);
  u += 0x7FFFu + ((u >> 16) & 1u);     // round-to-nearest-even
  return (bfr)(u >> 16);
}

__device__ __forceinline__ float actf(float x, int act) {
  if (act == 1) return fmaxf(x, 0.f);           // relu
  if (act == 2) {                                // softplus
    if (x > 20.f)  return x;
    if (x < -20.f) return __expf(x);
    return __logf(1.f + __expf(x));
  }
  return x;
}

// ---------------------------------------------------------------------------
// Fused up-to-3-layer MLP over 16-row tiles, bf16 WMMA 16x16x32.
// Input assembled from up to 4 (optionally gathered) f32 segments into LDS.
// A-frag (16x32 bf16): lane&15 = M, elems 0..15 -> K = ko + (lane>>4)*16 + e
// (same K order as B).
// B-frag (32x16): lane&15 = N column = weight row (W is [N x Kpad], K-major),
// elems 0..15 -> K = ko + (lane>>4)*16 + e.
// C/D: lane&15 = N, vgpr v -> M = v + (lane>>4)*8.
// ---------------------------------------------------------------------------
template <int MAXW>
__global__ __launch_bounds__(32) void mlp_fused_kernel(MlpArgs a) {
  __shared__ __align__(32) bfr sb0[16 * MAXW];
  __shared__ __align__(32) bfr sb1[16 * MAXW];
  const int lane = threadIdx.x;
  const int lr   = lane & 15;
  const int hi   = lane >> 4;
  const long long row0 = (long long)blockIdx.x * 16;

  int off[5];
  off[0] = 0;
  for (int i = 0; i < 4; ++i) off[i + 1] = off[i] + ((i < a.nseg) ? a.s[i].w : 0);

  // Stage gathered/concatenated input tile as bf16 (zero-padded).
  const int tot = 16 * a.K0pad;
  for (int j = lane; j < tot; j += 32) {
    int r  = j / a.K0pad;
    int cc = j - r * a.K0pad;
    long long row = row0 + r;
    float v = 0.f;
    if (row < a.M && cc < a.K0) {
      int si = 0;
      while (si + 1 < a.nseg && cc >= off[si + 1]) ++si;
      const Seg& S = a.s[si];
      long long g = S.idx ? (long long)S.idx[row] : row;
      v = S.p[g * (long long)S.w + (cc - off[si])];
    }
    sb0[j] = f2bf(v);
  }
  __syncthreads();

  bfr* cur = sb0;
  bfr* nxt = sb1;
  int Kpad = a.K0pad;
  for (int l = 0; l < a.nl; ++l) {
    const Layer L = a.L[l];
    const bool last = (l == a.nl - 1);
    for (int n0 = 0; n0 < L.N; n0 += 16) {
      v8f acc = {};
      for (int ko = 0; ko < Kpad; ko += 32) {
        const bfr* ap = cur + lr * Kpad + ko + hi * 16;
        v8bf alo = *(const v8bf*)ap;
        v8bf ahi = *(const v8bf*)(ap + 8);
        v16bf amat = __builtin_shufflevector(
            alo, ahi, 0, 1, 2, 3, 4, 5, 6, 7, 8, 9, 10, 11, 12, 13, 14, 15);
        v16bf bmat = *(const v16bf*)(L.W + (size_t)(n0 + lr) * Kpad + ko + hi * 16);
        acc = __builtin_amdgcn_wmma_f32_16x16x32_bf16(false, amat, false, bmat,
                                                      (short)0, acc, false, false);
      }
      const int col  = n0 + lr;
      const float bias = (col < L.N) ? L.b[col] : 0.f;
#pragma unroll
      for (int v = 0; v < 8; ++v) {
        const int m = v + hi * 8;
        float val = actf(acc[v] + bias, L.act);
        if (last) {
          long long row = row0 + m;
          if (row < a.M && col < L.N)
            a.out[row * (long long)a.ld_out + col] = val;
        } else if (col < L.N) {
          nxt[m * L.N + col] = f2bf(val);
        }
      }
    }
    __syncthreads();
    if (!last) { bfr* t = cur; cur = nxt; nxt = t; Kpad = L.N; }
  }
}

// -------------------------- small utility kernels --------------------------
__global__ void convw_kernel(const float* W, bfr* out, int N, int K, int Np, int Kp) {
  long long i = (long long)blockIdx.x * blockDim.x + threadIdx.x;
  long long tot = (long long)Np * Kp;
  if (i >= tot) return;
  int n = (int)(i / Kp), k = (int)(i - (long long)n * Kp);
  float v = (n < N && k < K) ? W[(size_t)n * K + k] : 0.f;
  out[i] = f2bf(v);
}

__global__ void fill_f32_kernel(float* p, float v, long long n) {
  long long i = (long long)blockIdx.x * blockDim.x + threadIdx.x;
  if (i < n) p[i] = v;
}
__global__ void fill_u32_kernel(unsigned int* p, unsigned int v, long long n) {
  long long i = (long long)blockIdx.x * blockDim.x + threadIdx.x;
  if (i < n) p[i] = v;
}
__global__ void copy_f32_kernel(float* d, const float* s, long long n) {
  long long i = (long long)blockIdx.x * blockDim.x + threadIdx.x;
  if (i < n) d[i] = s[i];
}
__global__ void ebm_kernel(const int* src, const int* batch, int* ebm, long long n) {
  long long i = (long long)blockIdx.x * blockDim.x + threadIdx.x;
  if (i < n) ebm[i] = batch[src[i]];
}
__global__ void seg_count_kernel(const int* seg, float* cnt, long long n) {
  long long i = (long long)blockIdx.x * blockDim.x + threadIdx.x;
  if (i < n) atomicAdd(&cnt[seg[i]], 1.f);
}
__global__ void scatter_sum_kernel(const float* vals, const int* seg, float* sums,
                                   long long n, int F) {
  long long t = (long long)blockIdx.x * blockDim.x + threadIdx.x;
  if (t >= n * F) return;
  long long i = t / F;
  int f = (int)(t - i * F);
  atomicAdd(&sums[(long long)seg[i] * F + f], vals[i * F + f]);
}
__global__ void mean_div_kernel(float* sums, const float* cnt, long long nseg, int F) {
  long long t = (long long)blockIdx.x * blockDim.x + threadIdx.x;
  if (t >= nseg * F) return;
  sums[t] = sums[t] / fmaxf(cnt[t / F], 1.f);
}

// --------------------------- Set2Set components ----------------------------
__global__ void lstm_kernel(const float* qs, const float* Wih, const float* bih,
                            const float* Whh, const float* bhh,
                            const float* h_in, float* h_out, float* c,
                            int G, int d) {
  int t = blockIdx.x * blockDim.x + threadIdx.x;
  if (t >= G * d) return;
  int g = t / d, j = t - (t / d) * d;
  const float* qg = qs + (size_t)g * 2 * d;
  const float* hg = h_in + (size_t)g * d;
  float gate[4];
#pragma unroll
  for (int k = 0; k < 4; ++k) {
    int row = k * d + j;
    float acc = bih[row] + bhh[row];
    const float* wi = Wih + (size_t)row * 2 * d;
    for (int q = 0; q < 2 * d; ++q) acc += qg[q] * wi[q];
    const float* wh = Whh + (size_t)row * d;
    for (int q = 0; q < d; ++q) acc += hg[q] * wh[q];
    gate[k] = acc;
  }
  float si = 1.f / (1.f + __expf(-gate[0]));
  float sf = 1.f / (1.f + __expf(-gate[1]));
  float tg = tanhf(gate[2]);
  float so = 1.f / (1.f + __expf(-gate[3]));
  float cn = sf * c[t] + si * tg;
  c[t] = cn;
  h_out[t] = so * tanhf(cn);
}

__device__ __forceinline__ unsigned int fkey_enc(float f) {
  unsigned int u = __float_as_uint(f);
  return (u & 0x80000000u) ? ~u : (u | 0x80000000u);
}
__device__ __forceinline__ float fkey_dec(unsigned int k) {
  unsigned int u = (k & 0x80000000u) ? (k ^ 0x80000000u) : ~k;
  return __uint_as_float(u);
}
__global__ void attn_dot_kernel(const float* xf, const int* seg, const float* h,
                                float* e_it, long long n, int d) {
  long long i = (long long)blockIdx.x * blockDim.x + threadIdx.x;
  if (i >= n) return;
  const float* xi = xf + i * d;
  const float* hg = h + (long long)seg[i] * d;
  float s = 0.f;
  for (int j = 0; j < d; ++j) s += xi[j] * hg[j];
  e_it[i] = s;
}
__global__ void seg_max_kernel(const float* e_it, const int* seg, unsigned int* mkey,
                               long long n) {
  long long i = (long long)blockIdx.x * blockDim.x + threadIdx.x;
  if (i < n) atomicMax(&mkey[seg[i]], fkey_enc(e_it[i]));
}
__global__ void exp_sum_kernel(const float* e_it, const int* seg,
                               const unsigned int* mkey, float* ssum, float* a_it,
                               long long n) {
  long long i = (long long)blockIdx.x * blockDim.x + threadIdx.x;
  if (i >= n) return;
  int g = seg[i];
  float ex = __expf(e_it[i] - fkey_dec(mkey[g]));
  a_it[i] = ex;
  atomicAdd(&ssum[g], ex);
}
__global__ void attn_scatter_kernel(const float* xf, const int* seg, const float* a_it,
                                    const float* ssum, float* r, long long n, int d) {
  long long t = (long long)blockIdx.x * blockDim.x + threadIdx.x;
  if (t >= n * d) return;
  long long i = t / d;
  int f = (int)(t - i * d);
  int g = seg[i];
  float w = a_it[i] / ssum[g];
  atomicAdd(&r[(long long)g * d + f], w * xf[i * d + f]);
}
__global__ void qstar_kernel(const float* h, const float* r, float* qs, int G, int d) {
  int t = blockIdx.x * blockDim.x + threadIdx.x;
  if (t >= G * 2 * d) return;
  int g = t / (2 * d), j = t - g * 2 * d;
  qs[t] = (j < d) ? h[g * d + j] : r[g * d + (j - d)];
}

}  // namespace

// ---------------------------------------------------------------------------
// Host orchestration.
// d_in layout assumption: [0]=x, [1]=edge_index, [2]=edge_attr,
// [3]=global_feats, [4]=batch, [5..84]=params leaves in JAX pytree order
// (dict keys sorted): blocks[0..1]{edge,glob,node}[0..2]{W,b}, dense_e,
// dense_g, dense_n (each [depth][lin]{W,b}), embed_e{W,b}, embed_g, embed_n,
// mlp[0..2]{W,b}, s2s_e{Whh,Wih,bhh,bih}, s2s_n{Whh,Wih,bhh,bih}.
// ---------------------------------------------------------------------------
extern "C" void kernel_launch(void* const* d_in, const int* in_sizes, int n_in,
                              void* d_out, int out_size, void* d_ws, size_t ws_size,
                              hipStream_t stream) {
  (void)in_sizes; (void)out_size; (void)ws_size;
  if (n_in < 85 || !d_ws) return;

  const float* x     = (const float*)d_in[0];
  const int*   ei    = (const int*)d_in[1];
  const int*   src   = ei;
  const int*   dst   = ei + kEdges;
  const float* ea    = (const float*)d_in[2];
  const float* gf    = (const float*)d_in[3];
  const int*   batch = (const int*)d_in[4];

  auto P = [&](int i) -> const float* { return (const float*)d_in[i]; };
  enum { P_DE = 41, P_DG = 49, P_DN = 57, P_EE = 65, P_EG = 67, P_EN = 69,
         P_MLP = 71, P_SE = 77, P_SN = 81 };

  char* wsb = (char*)d_ws;
  size_t off = 0;
  auto balloc = [&](size_t bytes) -> void* {
    off = (off + 255) & ~(size_t)255;
    void* r = wsb + off;
    off += bytes;
    return r;
  };
  auto G1 = [](long long n) { return dim3((unsigned)((n + 255) / 256)); };

  // bf16 weight arena (zero-padded: N->mult16, K->mult32).
  auto convW = [&](int widx, int N, int K) -> const bfr* {
    int Np = (N + 15) & ~15, Kp = (K + 31) & ~31;
    bfr* w = (bfr*)balloc((size_t)Np * Kp * sizeof(bfr));
    convw_kernel<<<G1((long long)Np * Kp), 256, 0, stream>>>(P(widx), w, N, K, Np, Kp);
    return w;
  };

  const bfr* w_en = convW(P_EN, kNH, 40);
  const bfr* w_ee = convW(P_EE, kEH, 12);
  const bfr* w_eg = convW(P_EG, kGH, 1);
  const bfr *w_dn0[2], *w_dn1[2], *w_de0[2], *w_de1[2], *w_dg0[2], *w_dg1[2];
  const bfr *w_be[2][3], *w_bn[2][3], *w_bg[2][3];
  for (int i = 0; i < 2; ++i) {
    w_dn0[i] = convW(P_DN + i * 4 + 0, 2 * kNH, kNH);
    w_dn1[i] = convW(P_DN + i * 4 + 2, kNH, 2 * kNH);
    w_de0[i] = convW(P_DE + i * 4 + 0, 2 * kEH, kEH);
    w_de1[i] = convW(P_DE + i * 4 + 2, kEH, 2 * kEH);
    w_dg0[i] = convW(P_DG + i * 4 + 0, 2 * kGH, kGH);
    w_dg1[i] = convW(P_DG + i * 4 + 2, kGH, 2 * kGH);
    int b0 = 5 + i * 18;
    w_be[i][0] = convW(b0 + 0, kEH, 2 * kNH + kEH + kGH);
    w_be[i][1] = convW(b0 + 2, kEH, kEH);
    w_be[i][2] = convW(b0 + 4, kEH, kEH);
    w_bg[i][0] = convW(b0 + 6, kGH, kNH + kEH + kGH);
    w_bg[i][1] = convW(b0 + 8, kGH, kGH);
    w_bg[i][2] = convW(b0 + 10, kGH, kGH);
    w_bn[i][0] = convW(b0 + 12, kNH, kNH + kEH + kGH);
    w_bn[i][1] = convW(b0 + 14, kNH, kNH);
    w_bn[i][2] = convW(b0 + 16, kNH, kNH);
  }
  const bfr* w_m0 = convW(P_MLP + 0, 512, 288);
  const bfr* w_m1 = convW(P_MLP + 2, 256, 512);
  const bfr* w_m2 = convW(P_MLP + 4, 1, 256);

  // Activation / scratch buffers.
  float* h_n     = (float*)balloc((size_t)kNodes * kNH * 4);
  float* h_e     = (float*)balloc((size_t)kEdges * kEH * 4);
  float* h_u     = (float*)balloc((size_t)kGraphs * kGH * 4);
  int*   ebm     = (int*)balloc((size_t)kEdges * 4);
  float* e_mean  = (float*)balloc((size_t)kNodes * kEH * 4);
  float* cnt_dst = (float*)balloc((size_t)kNodes * 4);
  float* em      = (float*)balloc((size_t)kGraphs * kEH * 4);
  float* nm      = (float*)balloc((size_t)kGraphs * kNH * 4);
  float* cnt_ebm = (float*)balloc((size_t)kGraphs * 4);
  float* cnt_bat = (float*)balloc((size_t)kGraphs * 4);
  float* qsn     = (float*)balloc((size_t)kGraphs * 2 * kNH * 4);
  float* qse     = (float*)balloc((size_t)kGraphs * 2 * kEH * 4);
  float* hA      = (float*)balloc((size_t)kGraphs * 64 * 4);
  float* hB      = (float*)balloc((size_t)kGraphs * 64 * 4);
  float* cbuf    = (float*)balloc((size_t)kGraphs * 64 * 4);
  float* rbuf    = (float*)balloc((size_t)kGraphs * 64 * 4);
  float* e_it    = (float*)balloc((size_t)kEdges * 4);
  float* a_it    = (float*)balloc((size_t)kEdges * 4);
  unsigned int* mkey = (unsigned int*)balloc((size_t)kGraphs * 4);
  float* ssum    = (float*)balloc((size_t)kGraphs * 4);

  auto mlp256 = [&](const MlpArgs& A) {
    mlp_fused_kernel<256><<<dim3((unsigned)((A.M + 15) / 16)), dim3(32), 0, stream>>>(A);
  };

  // Static graph structure (constant across depths).
  ebm_kernel<<<G1(kEdges), 256, 0, stream>>>(src, batch, ebm, kEdges);
  fill_f32_kernel<<<G1(kNodes), 256, 0, stream>>>(cnt_dst, 0.f, kNodes);
  seg_count_kernel<<<G1(kEdges), 256, 0, stream>>>(dst, cnt_dst, kEdges);
  fill_f32_kernel<<<G1(kGraphs), 256, 0, stream>>>(cnt_ebm, 0.f, kGraphs);
  seg_count_kernel<<<G1(kEdges), 256, 0, stream>>>(ebm, cnt_ebm, kEdges);
  fill_f32_kernel<<<G1(kGraphs), 256, 0, stream>>>(cnt_bat, 0.f, kGraphs);
  seg_count_kernel<<<G1(kNodes), 256, 0, stream>>>(batch, cnt_bat, kNodes);

  // Embeddings.
  { MlpArgs A{}; A.s[0] = {x, nullptr, 40}; A.nseg = 1; A.K0 = 40; A.K0pad = 64;
    A.L[0] = {w_en, P(P_EN + 1), kNH, 0}; A.nl = 1;
    A.M = kNodes; A.out = h_n; A.ld_out = kNH; mlp256(A); }
  { MlpArgs A{}; A.s[0] = {ea, nullptr, 12}; A.nseg = 1; A.K0 = 12; A.K0pad = 32;
    A.L[0] = {w_ee, P(P_EE + 1), kEH, 0}; A.nl = 1;
    A.M = kEdges; A.out = h_e; A.ld_out = kEH; mlp256(A); }
  { MlpArgs A{}; A.s[0] = {gf, nullptr, 1}; A.nseg = 1; A.K0 = 1; A.K0pad = 32;
    A.L[0] = {w_eg, P(P_EG + 1), kGH, 0}; A.nl = 1;
    A.M = kGraphs; A.out = h_u; A.ld_out = kGH; mlp256(A); }

  for (int dep = 0; dep < 2; ++dep) {
    const int b0 = 5 + dep * 18;
    // dense_n: Linear-ReLU-Linear (in place, row-local)
    { MlpArgs A{}; A.s[0] = {h_n, nullptr, kNH}; A.nseg = 1; A.K0 = kNH; A.K0pad = 64;
      A.L[0] = {w_dn0[dep], P(P_DN + dep * 4 + 1), 2 * kNH, 1};
      A.L[1] = {w_dn1[dep], P(P_DN + dep * 4 + 3), kNH, 0};
      A.nl = 2; A.M = kNodes; A.out = h_n; A.ld_out = kNH; mlp256(A); }
    // dense_e: Linear-Linear
    { MlpArgs A{}; A.s[0] = {h_e, nullptr, kEH}; A.nseg = 1; A.K0 = kEH; A.K0pad = 64;
      A.L[0] = {w_de0[dep], P(P_DE + dep * 4 + 1), 2 * kEH, 0};
      A.L[1] = {w_de1[dep], P(P_DE + dep * 4 + 3), kEH, 0};
      A.nl = 2; A.M = kEdges; A.out = h_e; A.ld_out = kEH; mlp256(A); }
    // dense_g
    { MlpArgs A{}; A.s[0] = {h_u, nullptr, kGH}; A.nseg = 1; A.K0 = kGH; A.K0pad = 32;
      A.L[0] = {w_dg0[dep], P(P_DG + dep * 4 + 1), 2 * kGH, 0};
      A.L[1] = {w_dg1[dep], P(P_DG + dep * 4 + 3), kGH, 0};
      A.nl = 2; A.M = kGraphs; A.out = h_u; A.ld_out = kGH; mlp256(A); }
    // edge update: [h_n[src] | h_n[dst] | h_e | h_u[ebm]] -> 64 sp -> 64 sp -> 64
    { MlpArgs A{};
      A.s[0] = {h_n, src, kNH}; A.s[1] = {h_n, dst, kNH};
      A.s[2] = {h_e, nullptr, kEH}; A.s[3] = {h_u, ebm, kGH};
      A.nseg = 4; A.K0 = 224; A.K0pad = 224;
      A.L[0] = {w_be[dep][0], P(b0 + 1), kEH, 2};
      A.L[1] = {w_be[dep][1], P(b0 + 3), kEH, 2};
      A.L[2] = {w_be[dep][2], P(b0 + 5), kEH, 0};
      A.nl = 3; A.M = kEdges; A.out = h_e; A.ld_out = kEH; mlp256(A); }
    // e_mean = seg_mean(h_e, dst, N_NODES)
    fill_f32_kernel<<<G1((long long)kNodes * kEH), 256, 0, stream>>>(
        e_mean, 0.f, (long long)kNodes * kEH);
    scatter_sum_kernel<<<G1((long long)kEdges * kEH), 256, 0, stream>>>(
        h_e, dst, e_mean, kEdges, kEH);
    mean_div_kernel<<<G1((long long)kNodes * kEH), 256, 0, stream>>>(
        e_mean, cnt_dst, kNodes, kEH);
    // node update: [h_n | e_mean | h_u[batch]] -> 64 sp -> 64 sp -> 64
    { MlpArgs A{};
      A.s[0] = {h_n, nullptr, kNH}; A.s[1] = {e_mean, nullptr, kEH};
      A.s[2] = {h_u, batch, kGH};
      A.nseg = 3; A.K0 = 160; A.K0pad = 160;
      A.L[0] = {w_bn[dep][0], P(b0 + 13), kNH, 2};
      A.L[1] = {w_bn[dep][1], P(b0 + 15), kNH, 2};
      A.L[2] = {w_bn[dep][2], P(b0 + 17), kNH, 0};
      A.nl = 3; A.M = kNodes; A.out = h_n; A.ld_out = kNH; mlp256(A); }
    // em / nm graph means
    fill_f32_kernel<<<G1((long long)kGraphs * kEH), 256, 0, stream>>>(
        em, 0.f, (long long)kGraphs * kEH);
    scatter_sum_kernel<<<G1((long long)kEdges * kEH), 256, 0, stream>>>(
        h_e, ebm, em, kEdges, kEH);
    mean_div_kernel<<<G1((long long)kGraphs * kEH), 256, 0, stream>>>(
        em, cnt_ebm, kGraphs, kEH);
    fill_f32_kernel<<<G1((long long)kGraphs * kNH), 256, 0, stream>>>(
        nm, 0.f, (long long)kGraphs * kNH);
    scatter_sum_kernel<<<G1((long long)kNodes * kNH), 256, 0, stream>>>(
        h_n, batch, nm, kNodes, kNH);
    mean_div_kernel<<<G1((long long)kGraphs * kNH), 256, 0, stream>>>(
        nm, cnt_bat, kGraphs, kNH);
    // global update: [em | nm | h_u] -> 32 sp -> 32 sp -> 32
    { MlpArgs A{};
      A.s[0] = {em, nullptr, kEH}; A.s[1] = {nm, nullptr, kNH};
      A.s[2] = {h_u, nullptr, kGH};
      A.nseg = 3; A.K0 = 160; A.K0pad = 160;
      A.L[0] = {w_bg[dep][0], P(b0 + 7), kGH, 2};
      A.L[1] = {w_bg[dep][1], P(b0 + 9), kGH, 2};
      A.L[2] = {w_bg[dep][2], P(b0 + 11), kGH, 0};
      A.nl = 3; A.M = kGraphs; A.out = h_u; A.ld_out = kGH; mlp256(A); }
  }

  // Set2Set readouts (d=64, 3 steps).
  auto set2set = [&](const float* xf, const int* seg, long long nit, int pW, float* qs) {
    const float* Whh = P(pW + 0);
    const float* Wih = P(pW + 1);
    const float* bhh = P(pW + 2);
    const float* bih = P(pW + 3);
    const int d = 64;
    fill_f32_kernel<<<G1((long long)kGraphs * 2 * d), 256, 0, stream>>>(
        qs, 0.f, (long long)kGraphs * 2 * d);
    fill_f32_kernel<<<G1((long long)kGraphs * d), 256, 0, stream>>>(
        hA, 0.f, (long long)kGraphs * d);
    fill_f32_kernel<<<G1((long long)kGraphs * d), 256, 0, stream>>>(
        cbuf, 0.f, (long long)kGraphs * d);
    for (int step = 0; step < 3; ++step) {
      lstm_kernel<<<G1((long long)kGraphs * d), 256, 0, stream>>>(
          qs, Wih, bih, Whh, bhh, hA, hB, cbuf, kGraphs, d);
      copy_f32_kernel<<<G1((long long)kGraphs * d), 256, 0, stream>>>(
          hA, hB, (long long)kGraphs * d);
      fill_u32_kernel<<<G1(kGraphs), 256, 0, stream>>>(mkey, 0x00800000u, kGraphs);
      fill_f32_kernel<<<G1(kGraphs), 256, 0, stream>>>(ssum, 0.f, kGraphs);
      fill_f32_kernel<<<G1((long long)kGraphs * d), 256, 0, stream>>>(
          rbuf, 0.f, (long long)kGraphs * d);
      attn_dot_kernel<<<G1(nit), 256, 0, stream>>>(xf, seg, hA, e_it, nit, d);
      seg_max_kernel<<<G1(nit), 256, 0, stream>>>(e_it, seg, mkey, nit);
      exp_sum_kernel<<<G1(nit), 256, 0, stream>>>(e_it, seg, mkey, ssum, a_it, nit);
      attn_scatter_kernel<<<G1(nit * d), 256, 0, stream>>>(xf, seg, a_it, ssum, rbuf, nit, d);
      qstar_kernel<<<G1((long long)kGraphs * 2 * d), 256, 0, stream>>>(hA, rbuf, qs, kGraphs, d);
    }
  };
  set2set(h_n, batch, kNodes, P_SN, qsn);
  set2set(h_e, ebm, kEdges, P_SE, qse);

  // Final MLP: [qsn | qse | h_u] (288) -> 512 relu -> 256 relu -> 1
  { MlpArgs A{};
    A.s[0] = {qsn, nullptr, 128}; A.s[1] = {qse, nullptr, 128};
    A.s[2] = {h_u, nullptr, kGH};
    A.nseg = 3; A.K0 = 288; A.K0pad = 288;
    A.L[0] = {w_m0, P(P_MLP + 1), 512, 1};
    A.L[1] = {w_m1, P(P_MLP + 3), 256, 1};
    A.L[2] = {w_m2, P(P_MLP + 5), 1, 0};
    A.nl = 3; A.M = kGraphs; A.out = (float*)d_out; A.ld_out = 1;
    mlp_fused_kernel<512><<<dim3((unsigned)((kGraphs + 15) / 16)), dim3(32), 0, stream>>>(A);
  }
}
